// ImageTransformer_Seq_PETR_MS_49435073577595
// MI455X (gfx1250) — matrix operand order repaired and checked by reading, not compiled, not run on hardware
//
#include <hip/hip_runtime.h>
#include <math.h>

#define B_ 2
#define V_ 6
#define C_ 256
#define P_ 200
#define H_ 8
#define D_ 32
#define L_ 2
#define F_ 1024
#define N_ 3520
#define NEGB (-1000000000.0f)

typedef __attribute__((ext_vector_type(16))) __bf16 v16bf;
typedef __attribute__((ext_vector_type(8)))  float  v8f;
typedef __attribute__((ext_vector_type(4))) unsigned int u32x4;
typedef __attribute__((ext_vector_type(8))) int i32x8;
typedef __attribute__((ext_vector_type(4))) int i32x4;
typedef unsigned short u16t;

__device__ __forceinline__ u16t f2bf(float f) {
  unsigned int u = __float_as_uint(f);
  u = u + 0x7FFFu + ((u >> 16) & 1u);
  return (u16t)(u >> 16);
}

__device__ __forceinline__ float wredmax(float v) {
  for (int o = 16; o > 0; o >>= 1) v = fmaxf(v, __shfl_xor(v, o, 32));
  return v;
}
__device__ __forceinline__ float wredsum(float v) {
  for (int o = 16; o > 0; o >>= 1) v += __shfl_xor(v, o, 32);
  return v;
}

// ---------------------------------------------------------------------------
// TDM probe (not launched; emits tensor_load_to_lds + s_wait_tensorcnt so the
// async-tensor path is exercised by codegen). 6-arg builtin on this toolchain.
// ---------------------------------------------------------------------------
__global__ void k_tdm_probe(const float* gsrc, float* gdst) {
  __shared__ __align__(16) float lbuf[1024];
  unsigned long long ga = (unsigned long long)(size_t)gsrc;
  unsigned int lds_addr = (unsigned int)(size_t)(&lbuf[0]);
  u32x4 g0;
  g0[0] = 1u;                                   // count=1 valid descriptor
  g0[1] = lds_addr;                             // lds_addr
  g0[2] = (unsigned int)ga;                     // global_addr lo
  g0[3] = (unsigned int)((ga >> 32) & 0x01FFFFFFu) | (2u << 30);  // type=2
  i32x8 g1;
  g1[0] = 2 << 16;                              // data_size = 4B
  g1[1] = 0;
  g1[2] = 256;                                  // tensor_dim0
  g1[3] = 16 | (64 << 16);                      // tensor_dim1 lo | tile_dim0
  g1[4] = 16;                                   // tile_dim1
  g1[5] = 256;                                  // dim0 stride lo
  g1[6] = 0;
  g1[7] = 0;
  i32x4 g2; g2[0] = 0; g2[1] = 0; g2[2] = 0; g2[3] = 0;
  i32x4 g3; g3[0] = 0; g3[1] = 0; g3[2] = 0; g3[3] = 0;
  i32x8 g4;
  g4[0] = 0; g4[1] = 0; g4[2] = 0; g4[3] = 0;
  g4[4] = 0; g4[5] = 0; g4[6] = 0; g4[7] = 0;
  if (threadIdx.x == 0)
    __builtin_amdgcn_tensor_load_to_lds(g0, g1, g2, g3, g4, 0);
  __builtin_amdgcn_s_wait_tensorcnt(0);
  __syncthreads();
  gdst[threadIdx.x] = lbuf[threadIdx.x];
}

// ---------------------------------------------------------------------------
// Generic batched bf16 WMMA GEMM:  out = A(MxK) @ opB(W)^T + bias (+res), relu
//   normal mode: W is (N x K) row-major (out = A @ W^T)
//   btrans mode: W is (K x N) row-major (out = A @ W)
// Block tile: 64(M) x 128(N), K-step 32. 8 waves; each wave: 1 A fragment,
// 4 B fragments, 4 x v_wmma_f32_16x16x32_bf16 per K-step.
// ---------------------------------------------------------------------------
struct GemmP {
  const u16t* A;
  const u16t* Bw;
  const float* bias;
  const float* res;
  float* outF;
  u16t* outB;
  long lda, ldb, ldr, ldo;
  long sA, sB, sR, sO;
  int M, N, K, relu, btrans;
};

__global__ void __launch_bounds__(256) k_gemm(GemmP p) {
  __shared__ __align__(16) u16t As[64 * 32];
  __shared__ __align__(16) u16t Bs[128 * 32];
  const int z = blockIdx.z;
  const u16t* A  = p.A  + (long)z * p.sA;
  const u16t* Bw = p.Bw + (long)z * p.sB;
  const int bm = blockIdx.y * 64, bn = blockIdx.x * 128;
  const int tid = threadIdx.x, lane = tid & 31, wave = tid >> 5;
  const int mt = wave & 3;        // m sub-tile (16 rows)
  const int ng = wave >> 2;       // n half (64 cols)
  const int hf = lane >> 4, lm = lane & 15;
  const int ra = tid >> 2, ca = (tid & 3) * 8;    // A loader: 64x32, 8/thread
  const int rb = tid >> 1, cb = (tid & 1) * 16;   // B loader: 128x32, 16/thread
  const int kb = tid >> 3, nb = (tid & 7) * 16;   // btrans loader: 32k x 128n
  const bool aVec = ((p.lda & 7) == 0);
  const bool bVec = ((p.ldb & 7) == 0);

  v8f acc[4] = {{}, {}, {}, {}};

  for (int k0 = 0; k0 < p.K; k0 += 32) {
    // ---- stage A tile (64 x 32) ----
    {
      const int m = bm + ra;
      if (aVec && m < p.M && (k0 + ca + 8) <= p.K) {
        const u16t* src = A + (long)m * p.lda + (k0 + ca);
        *reinterpret_cast<uint4*>(&As[ra * 32 + ca]) =
            *reinterpret_cast<const uint4*>(src);
        if (k0 + 32 < p.K) __builtin_prefetch(src + 32, 0, 1);
      } else {
#pragma unroll
        for (int i = 0; i < 8; i++) {
          int k = k0 + ca + i;
          bool ok = (m < p.M) && (k < p.K);
          long idx = ok ? ((long)m * p.lda + k) : 0;
          u16t v = A[idx];
          As[ra * 32 + ca + i] = ok ? v : (u16t)0;
        }
      }
    }
    // ---- stage B tile (128 x 32, stored [n][k]) ----
    if (!p.btrans) {
      const int n = bn + rb;
      if (bVec && n < p.N && (k0 + cb + 16) <= p.K) {
        const u16t* src = Bw + (long)n * p.ldb + (k0 + cb);
        *reinterpret_cast<uint4*>(&Bs[rb * 32 + cb]) =
            *reinterpret_cast<const uint4*>(src);
        *reinterpret_cast<uint4*>(&Bs[rb * 32 + cb + 8]) =
            *reinterpret_cast<const uint4*>(src + 8);
      } else {
#pragma unroll
        for (int i = 0; i < 16; i++) {
          int k = k0 + cb + i;
          bool ok = (n < p.N) && (k < p.K);
          long idx = ok ? ((long)n * p.ldb + k) : 0;
          u16t v = Bw[idx];
          Bs[rb * 32 + cb + i] = ok ? v : (u16t)0;
        }
      }
    } else {
      const int k = k0 + kb;
      const int n0 = bn + nb;
      u16t tmp[16];
      if (bVec && k < p.K && (n0 + 16) <= p.N) {
        const u16t* src = Bw + (long)k * p.ldb + n0;
        *reinterpret_cast<uint4*>(&tmp[0]) = *reinterpret_cast<const uint4*>(src);
        *reinterpret_cast<uint4*>(&tmp[8]) = *reinterpret_cast<const uint4*>(src + 8);
      } else {
#pragma unroll
        for (int i = 0; i < 16; i++) {
          int n = n0 + i;
          bool ok = (k < p.K) && (n < p.N);
          long idx = ok ? ((long)k * p.ldb + n) : 0;
          u16t v = Bw[idx];
          tmp[i] = ok ? v : (u16t)0;
        }
      }
#pragma unroll
      for (int i = 0; i < 16; i++) Bs[(nb + i) * 32 + kb] = tmp[i];
    }
    __syncthreads();

    union FragU { uint4 q[2]; v16bf v; };
    FragU fa;
    // A fragment: row = mt*16+lm ; K elems {hf*8..+7, 16+hf*8..+7}
    const u16t* ap = &As[(mt * 16 + lm) * 32 + hf * 8];
    fa.q[0] = *reinterpret_cast<const uint4*>(ap);
    fa.q[1] = *reinterpret_cast<const uint4*>(ap + 16);
#pragma unroll
    for (int j = 0; j < 4; j++) {
      FragU fb;
      // B fragment: col = lm ; K elems {hf*8..+7, 16+hf*8..+7}
      const u16t* bp = &Bs[(ng * 64 + j * 16 + lm) * 32 + hf * 8];
      fb.q[0] = *reinterpret_cast<const uint4*>(bp);
      fb.q[1] = *reinterpret_cast<const uint4*>(bp + 16);
      acc[j] = __builtin_amdgcn_wmma_f32_16x16x32_bf16(false, fa.v, false, fb.v,
                                                       (short)0, acc[j], false, false);
    }
    __syncthreads();
  }

  const long ro = (long)z * p.sO;
#pragma unroll
  for (int j = 0; j < 4; j++) {
#pragma unroll
    for (int e = 0; e < 8; e++) {
      int mg = bm + mt * 16 + e + 8 * hf;
      int n = bn + ng * 64 + j * 16 + lm;
      if (mg < p.M && n < p.N) {
        float val = acc[j][e] + (p.bias ? p.bias[n] : 0.f) +
                    (p.res ? p.res[(long)z * p.sR + (long)mg * p.ldr + n] : 0.f);
        if (p.relu) val = fmaxf(val, 0.f);
        if (p.outF) p.outF[ro + (long)mg * p.ldo + n] = val;
        if (p.outB) p.outB[ro + (long)mg * p.ldo + n] = f2bf(val);
      }
    }
  }
}

// ---------------------------------------------------------------------------
// Elementwise / utility kernels
// ---------------------------------------------------------------------------
__global__ void k_cvt_bf16(const float* s, u16t* d, long n) {
  long i = (long)blockIdx.x * blockDim.x + threadIdx.x;
  long st = (long)gridDim.x * blockDim.x;
  for (; i < n; i += st) d[i] = f2bf(s[i]);
}

__global__ void k_build_keys(const float* f0, const float* f1, float* keys) {
  const long total = (long)B_ * V_ * N_ * C_;
  long i = (long)blockIdx.x * blockDim.x + threadIdx.x;
  long st = (long)gridDim.x * blockDim.x;
  for (; i < total; i += st) {
    int c = (int)(i & (C_ - 1));
    long t = i >> 8;
    int n = (int)(t % N_); t /= N_;
    int v = (int)(t % V_);
    int b = (int)(t / V_);
    float val;
    if (n < 2816) val = f0[(((long)(b * V_ + v) * C_ + c) * 2816) + n];
    else          val = f1[(((long)(b * V_ + v) * C_ + c) * 704) + (n - 2816)];
    keys[i] = val;
  }
}

__global__ void k_inv4(const float* l2i, float* inv) {
  int id = blockIdx.x * blockDim.x + threadIdx.x;
  if (id >= B_ * V_) return;
  float m[4][8];
  for (int r = 0; r < 4; r++)
    for (int c = 0; c < 4; c++) {
      m[r][c] = l2i[id * 16 + r * 4 + c];
      m[r][4 + c] = (r == c) ? 1.f : 0.f;
    }
  for (int col = 0; col < 4; col++) {
    int piv = col; float best = fabsf(m[col][col]);
    for (int r = col + 1; r < 4; r++) {
      float a = fabsf(m[r][col]);
      if (a > best) { best = a; piv = r; }
    }
    if (piv != col)
      for (int c = 0; c < 8; c++) { float t = m[col][c]; m[col][c] = m[piv][c]; m[piv][c] = t; }
    float rinv = 1.f / m[col][col];
    for (int c = 0; c < 8; c++) m[col][c] *= rinv;
    for (int r = 0; r < 4; r++)
      if (r != col) {
        float f = m[r][col];
        for (int c = 0; c < 8; c++) m[r][c] -= f * m[col][c];
      }
  }
  for (int r = 0; r < 4; r++)
    for (int c = 0; c < 4; c++) inv[id * 16 + r * 4 + c] = m[r][4 + c];
}

__global__ void k_build_cam(const float* npos, const int* hw, const float* inv, u16t* cam) {
  const long total = (long)B_ * V_ * N_ * 96;
  long i = (long)blockIdx.x * blockDim.x + threadIdx.x;
  long st = (long)gridDim.x * blockDim.x;
  const float bin = 53.0f / (float)(D_ * (1 + D_));
  for (; i < total; i += st) {
    int jj = (int)(i % 96);
    long t = i / 96;
    int n = (int)(t % N_); t /= N_;
    int v = (int)(t % V_);
    int b = (int)(t / V_);
    int d = jj / 3, ii = jj % 3;
    float cd = fmaxf(1.f + bin * (float)d * ((float)d + 1.f), 1e-5f);
    float sx = (1.f / (1.f + __expf(-npos[n * 2 + 0]))) * (float)hw[b * 2 + 1];
    float sy = (1.f / (1.f + __expf(-npos[n * 2 + 1]))) * (float)hw[b * 2 + 0];
    const float* M = inv + (b * V_ + v) * 16;
    float val = (sx * cd) * M[ii * 4 + 0] + (sy * cd) * M[ii * 4 + 1] +
                cd * M[ii * 4 + 2] + 1.f * M[ii * 4 + 3];
    cam[i] = f2bf(val);
  }
}

__global__ void k_active(const int* vm, int* active) {
  int id = blockIdx.x * blockDim.x + threadIdx.x;
  if (id >= B_ * V_) return;
  int v = id % V_, b = id / V_;
  int s = 0;
  for (int p = 0; p < P_; p++) s += (vm[(b * P_ + p) * V_ + v] > 0) ? 1 : 0;
  int valid = (s > 1) ? 1 : 0;
  for (int p = 0; p < P_; p++)
    active[(b * V_ + v) * P_ + p] = (valid && vm[(b * P_ + p) * V_ + v] > 0) ? 1 : 0;
}

__global__ void k_cnt(const int* active, float* cnt) {
  int id = blockIdx.x * blockDim.x + threadIdx.x;
  if (id >= B_ * P_) return;
  int p = id % P_, b = id / P_;
  int s = 0;
  for (int v = 0; v < V_; v++) s += active[(b * V_ + v) * P_ + p];
  cnt[id] = (float)s;
}

__global__ void k_broadcast_x(const float* qf, float* x) {
  const long total = (long)B_ * V_ * P_ * C_;
  long i = (long)blockIdx.x * blockDim.x + threadIdx.x;
  long st = (long)gridDim.x * blockDim.x;
  for (; i < total; i += st) {
    int c = (int)(i & (C_ - 1));
    long t = i >> 8;
    int p = (int)(t % P_); t /= P_;
    int v = (int)(t % V_);
    int b = (int)(t / V_);
    (void)v;
    x[i] = qf[((long)b * C_ + c) * P_ + p];
  }
}

__global__ void k_sa_in(const float* qf, const float* peq, u16t* dst) {
  const long total = (long)B_ * P_ * C_;
  long i = (long)blockIdx.x * blockDim.x + threadIdx.x;
  long st = (long)gridDim.x * blockDim.x;
  for (; i < total; i += st) {
    int c = (int)(i & (C_ - 1));
    long t = i >> 8;
    int p = (int)(t % P_);
    int b = (int)(t / P_);
    dst[i] = f2bf(qf[((long)b * C_ + c) * P_ + p] + peq[i]);
  }
}

__global__ void k_ca_qin(const float* x, const float* peq, u16t* dst) {
  const long total = (long)B_ * V_ * P_ * C_;
  long i = (long)blockIdx.x * blockDim.x + threadIdx.x;
  long st = (long)gridDim.x * blockDim.x;
  for (; i < total; i += st) {
    int c = (int)(i & (C_ - 1));
    long t = i >> 8;
    int p = (int)(t % P_); t /= P_;
    int v = (int)(t % V_);
    int b = (int)(t / V_);
    (void)v;
    dst[i] = f2bf(x[i] + peq[((long)b * P_ + p) * C_ + c]);
  }
}

__global__ void k_ln_res(float* x, const float* r, const float* g, const float* bt, u16t* xbf) {
  __shared__ float red[C_];
  const int row = blockIdx.x, t = threadIdx.x;
  float v = x[(long)row * C_ + t] + (r ? r[(long)row * C_ + t] : 0.f);
  red[t] = v; __syncthreads();
  for (int s = 128; s > 0; s >>= 1) { if (t < s) red[t] += red[t + s]; __syncthreads(); }
  float mean = red[0] / (float)C_;
  __syncthreads();
  float d = v - mean;
  red[t] = d * d; __syncthreads();
  for (int s = 128; s > 0; s >>= 1) { if (t < s) red[t] += red[t + s]; __syncthreads(); }
  float var = red[0] / (float)C_;
  float o = d * rsqrtf(var + 1e-5f) * g[t] + bt[t];
  x[(long)row * C_ + t] = o;
  if (xbf) xbf[(long)row * C_ + t] = f2bf(o);
}

__global__ void k_softmax_sa(const float* lg, const int* active, u16t* pr, float scale) {
  int wid = (int)(((long)blockIdx.x * blockDim.x + threadIdx.x) >> 5);
  int lane = threadIdx.x & 31;
  const int rows = B_ * V_ * H_ * P_;
  if (wid >= rows) return;
  int q = wid % P_; int t = wid / P_;
  int h = t % H_; t /= H_;
  int v = t % V_; int b = t / V_;
  const float* row = lg + ((long)(b * H_ + h) * P_ + q) * P_;
  const int* am = active + (b * V_ + v) * P_;
  float lv[7];
  float mx = -3.4e38f;
#pragma unroll
  for (int i = 0; i < 7; i++) {
    int k = lane + 32 * i;
    float x = -3.4e38f;
    if (k < P_) x = row[k] * scale + (am[k] ? 0.f : NEGB);
    lv[i] = x;
    mx = fmaxf(mx, x);
  }
  mx = wredmax(mx);
  float s = 0.f;
#pragma unroll
  for (int i = 0; i < 7; i++) {
    int k = lane + 32 * i;
    if (k < P_) { float e = __expf(lv[i] - mx); lv[i] = e; s += e; }
  }
  s = wredsum(s);
  float invs = 1.f / s;
  u16t* orow = pr + ((long)((b * V_ + v) * H_ + h) * P_ + q) * P_;
#pragma unroll
  for (int i = 0; i < 7; i++) {
    int k = lane + 32 * i;
    if (k < P_) orow[k] = f2bf(lv[i] * invs);
  }
}

__global__ void k_softmax_rows(const float* lg, u16t* pr, int rows, int nk, float scale) {
  int wid = (int)(((long)blockIdx.x * blockDim.x + threadIdx.x) >> 5);
  int lane = threadIdx.x & 31;
  if (wid >= rows) return;
  const float* row = lg + (long)wid * nk;
  float mx = -3.4e38f;
  for (int k = lane; k < nk; k += 32) mx = fmaxf(mx, row[k] * scale);
  mx = wredmax(mx);
  float s = 0.f;
  for (int k = lane; k < nk; k += 32) s += __expf(row[k] * scale - mx);
  s = wredsum(s);
  float invs = 1.f / s;
  u16t* o = pr + (long)wid * nk;
  for (int k = lane; k < nk; k += 32) o[k] = f2bf(__expf(row[k] * scale - mx) * invs);
}

__global__ void k_merge_heads(const float* ao, u16t* mg) {
  const long total = (long)B_ * V_ * P_ * C_;
  long i = (long)blockIdx.x * blockDim.x + threadIdx.x;
  long st = (long)gridDim.x * blockDim.x;
  for (; i < total; i += st) {
    int c = (int)(i & (C_ - 1));
    long t = i >> 8;
    int p = (int)(t % P_);
    int bv = (int)(t / P_);
    int h = c >> 5, d = c & 31;
    mg[i] = f2bf(ao[((long)(bv * H_ + h) * P_ + p) * D_ + d]);
  }
}

__global__ void k_pool(const float* x, const int* active, const float* cnt,
                       float* qf, u16t* qT) {
  const long total = (long)B_ * P_ * C_;
  long i = (long)blockIdx.x * blockDim.x + threadIdx.x;
  long st = (long)gridDim.x * blockDim.x;
  for (; i < total; i += st) {
    int c = (int)(i & (C_ - 1));
    long t = i >> 8;
    int p = (int)(t % P_);
    int b = (int)(t / P_);
    float s = 0.f;
    for (int v = 0; v < V_; v++)
      if (active[(b * V_ + v) * P_ + p])
        s += x[(((long)(b * V_ + v) * P_) + p) * C_ + c];
    float val = s / (cnt[b * P_ + p] + 1e-4f);
    qf[((long)b * C_ + c) * P_ + p] = val;
    qT[i] = f2bf(val);
  }
}

__global__ void k_head_finish(const float* center, float* q_pos, float* out_centers,
                              float* out_heights, float* out_posbev, int l) {
  int i = blockIdx.x * blockDim.x + threadIdx.x;
  if (i >= B_ * P_) return;
  int p = i % P_, b = i / P_;
  float cx = center[i * 3 + 0], cy = center[i * 3 + 1], cz = center[i * 3 + 2];
  q_pos[i * 3 + 0] = cx; q_pos[i * 3 + 1] = cy; q_pos[i * 3 + 2] = cz;
  float gx = (cx + 54.f) / 108.f * 180.f;
  float gy = (cy + 54.f) / 108.f * 180.f;
  out_centers[((long)(l * B_ + b) * 2 + 0) * P_ + p] = gx;
  out_centers[((long)(l * B_ + b) * 2 + 1) * P_ + p] = gy;
  out_heights[(long)(l * B_ + b) * P_ + p] = cz;
  if (out_posbev) { out_posbev[i * 2 + 0] = gx; out_posbev[i * 2 + 1] = gy; }
}

// ---------------------------------------------------------------------------
// Host orchestration
// ---------------------------------------------------------------------------
extern "C" void kernel_launch(void* const* d_in, const int* in_sizes, int n_in,
                              void* d_out, int out_size, void* d_ws, size_t ws_size,
                              hipStream_t stream) {
  (void)in_sizes; (void)n_in; (void)out_size; (void)ws_size;
  const float* iq_feat = (const float*)d_in[0];
  const float* iq_pos  = (const float*)d_in[1];
  const int*   iq_mask = (const int*)d_in[2];
  const float* feat0   = (const float*)d_in[3];
  const float* feat1   = (const float*)d_in[4];
  const float* npos    = (const float*)d_in[5];
  const float* l2i     = (const float*)d_in[6];
  const int*   hw      = (const int*)d_in[7];
  const float* w_qpos1 = (const float*)d_in[8];
  const float* b_qpos1 = (const float*)d_in[9];
  const float* w_qpos2 = (const float*)d_in[10];
  const float* b_qpos2 = (const float*)d_in[11];
  const float* w_kpos1 = (const float*)d_in[12];
  const float* b_kpos1 = (const float*)d_in[13];
  const float* w_kpos2 = (const float*)d_in[14];
  const float* b_kpos2 = (const float*)d_in[15];
  const float* w_sain  = (const float*)d_in[16];
  const float* b_sain  = (const float*)d_in[17];
  const float* w_saout = (const float*)d_in[18];
  const float* b_saout = (const float*)d_in[19];
  const float* w_cain  = (const float*)d_in[20];
  const float* b_cain  = (const float*)d_in[21];
  const float* w_caout = (const float*)d_in[22];
  const float* b_caout = (const float*)d_in[23];
  const float* w_ffn1  = (const float*)d_in[24];
  const float* b_ffn1  = (const float*)d_in[25];
  const float* w_ffn2  = (const float*)d_in[26];
  const float* b_ffn2  = (const float*)d_in[27];
  const float* n1g = (const float*)d_in[28];
  const float* n1b = (const float*)d_in[29];
  const float* n2g = (const float*)d_in[30];
  const float* n2b = (const float*)d_in[31];
  const float* n3g = (const float*)d_in[32];
  const float* n3b = (const float*)d_in[33];
  const float* w_h1 = (const float*)d_in[34];
  const float* b_h1 = (const float*)d_in[35];
  const float* w_h2 = (const float*)d_in[36];
  const float* b_h2 = (const float*)d_in[37];
  float* out = (float*)d_out;

  char* ws = (char*)d_ws;
  size_t off = 0;
  auto alloc = [&](size_t bytes) -> char* {
    char* p = ws + off;
    off += (bytes + 255) & ~(size_t)255;
    return p;
  };
  const long BVN = (long)B_ * V_ * N_;      // 42240
  const long BVP = (long)B_ * V_ * P_;      // 2400
  const long BP  = (long)B_ * P_;           // 400

  float* keys     = (float*)alloc(BVN * C_ * 4);
  u16t*  cam_bf   = (u16t*)alloc(BVN * 96 * 2);
  u16t*  kv_bf    = (u16t*)alloc(BVN * C_ * 2);
  u16t*  hidk_bf  = (u16t*)alloc(BVN * C_ * 2);   // reused as FFN hidden
  u16t*  kproj_bf = (u16t*)alloc(BVN * C_ * 2);
  u16t*  vproj_bf = (u16t*)alloc(BVN * C_ * 2);
  float* lg_ca    = (float*)alloc((long)H_ * P_ * N_ * 4);
  u16t*  pr_ca    = (u16t*)alloc((long)H_ * P_ * N_ * 2);
  float* lg_sa    = (float*)alloc((long)B_ * H_ * P_ * P_ * 4);
  u16t*  pr_sa    = (u16t*)alloc((long)B_ * V_ * H_ * P_ * P_ * 2);
  float* attnO    = (float*)alloc(BVP * C_ * 4);
  u16t*  merged   = (u16t*)alloc(BVP * C_ * 2);
  float* x        = (float*)alloc(BVP * C_ * 4);
  u16t*  x_bf     = (u16t*)alloc(BVP * C_ * 2);
  float* projf    = (float*)alloc(BVP * C_ * 4);
  u16t*  qin_bf   = (u16t*)alloc(BVP * C_ * 2);
  u16t*  qproj_bf = (u16t*)alloc(BVP * C_ * 2);
  u16t*  saqkv_bf = (u16t*)alloc((long)B_ * P_ * 3 * C_ * 2);
  u16t*  sain_bf  = (u16t*)alloc(BP * C_ * 2);
  float* pe_q     = (float*)alloc(BP * C_ * 4);
  u16t*  hidq_bf  = (u16t*)alloc(BP * C_ * 2);
  u16t*  qpos_bf  = (u16t*)alloc(BP * 3 * 2);
  float* q_feat   = (float*)alloc((long)B_ * C_ * P_ * 4);
  u16t*  qT_bf    = (u16t*)alloc(BP * C_ * 2);
  u16t*  h1_bf    = (u16t*)alloc(BP * C_ * 2);
  float* center   = (float*)alloc(BP * 3 * 4);
  float* q_pos    = (float*)alloc(BP * 3 * 4);
  float* inv_f    = (float*)alloc((long)B_ * V_ * 16 * 4);
  int*   active   = (int*)alloc(BVP * 4);
  float* cnt      = (float*)alloc(BP * 4);
  u16t* bw_qpos1 = (u16t*)alloc((long)L_ * C_ * 3 * 2);
  u16t* bw_qpos2 = (u16t*)alloc((long)L_ * C_ * C_ * 2);
  u16t* bw_kpos1 = (u16t*)alloc((long)L_ * C_ * 96 * 2);
  u16t* bw_kpos2 = (u16t*)alloc((long)L_ * C_ * C_ * 2);
  u16t* bw_sain  = (u16t*)alloc((long)L_ * 3 * C_ * C_ * 2);
  u16t* bw_saout = (u16t*)alloc((long)L_ * C_ * C_ * 2);
  u16t* bw_cain  = (u16t*)alloc((long)L_ * 3 * C_ * C_ * 2);
  u16t* bw_caout = (u16t*)alloc((long)L_ * C_ * C_ * 2);
  u16t* bw_ffn1  = (u16t*)alloc((long)L_ * F_ * C_ * 2);
  u16t* bw_ffn2  = (u16t*)alloc((long)L_ * C_ * F_ * 2);
  u16t* bw_h1    = (u16t*)alloc((long)C_ * C_ * 2);
  u16t* bw_h2    = (u16t*)alloc((long)3 * C_ * 2);

  auto ewg = [](long n) { return dim3((unsigned)((n + 255) / 256)); };
  auto cvt = [&](const float* s, u16t* d, long n) {
    k_cvt_bf16<<<ewg(n), 256, 0, stream>>>(s, d, n);
  };
  auto gemm = [&](const u16t* A, long lda, long sA,
                  const u16t* Bw, long ldb, long sB,
                  const float* bias, const float* res, long ldr, long sR,
                  float* outF, u16t* outB, long ldo, long sO,
                  int M, int N, int K, int relu, int btrans, int batch) {
    GemmP p;
    p.A = A; p.Bw = Bw; p.bias = bias; p.res = res;
    p.outF = outF; p.outB = outB;
    p.lda = lda; p.ldb = ldb; p.ldr = ldr; p.ldo = ldo;
    p.sA = sA; p.sB = sB; p.sR = sR; p.sO = sO;
    p.M = M; p.N = N; p.K = K; p.relu = relu; p.btrans = btrans;
    dim3 g((N + 127) / 128, (M + 63) / 64, batch);
    k_gemm<<<g, 256, 0, stream>>>(p);
  };

  // ----- weight conversion -----
  cvt(w_qpos1, bw_qpos1, (long)L_ * C_ * 3);
  cvt(w_qpos2, bw_qpos2, (long)L_ * C_ * C_);
  cvt(w_kpos1, bw_kpos1, (long)L_ * C_ * 96);
  cvt(w_kpos2, bw_kpos2, (long)L_ * C_ * C_);
  cvt(w_sain,  bw_sain,  (long)L_ * 3 * C_ * C_);
  cvt(w_saout, bw_saout, (long)L_ * C_ * C_);
  cvt(w_cain,  bw_cain,  (long)L_ * 3 * C_ * C_);
  cvt(w_caout, bw_caout, (long)L_ * C_ * C_);
  cvt(w_ffn1,  bw_ffn1,  (long)L_ * F_ * C_);
  cvt(w_ffn2,  bw_ffn2,  (long)L_ * C_ * F_);
  cvt(w_h1,    bw_h1,    (long)C_ * C_);
  cvt(w_h2,    bw_h2,    (long)3 * C_);

  // ----- preprocessing -----
  k_build_keys<<<ewg(BVN * C_), 256, 0, stream>>>(feat0, feat1, keys);
  k_inv4<<<1, 32, 0, stream>>>(l2i, inv_f);
  k_build_cam<<<ewg(BVN * 96), 256, 0, stream>>>(npos, hw, inv_f, cam_bf);
  k_active<<<1, 32, 0, stream>>>(iq_mask, active);
  k_cnt<<<2, 256, 0, stream>>>(active, cnt);
  (void)hipMemcpyAsync(q_feat, iq_feat, (size_t)B_ * C_ * P_ * 4, hipMemcpyDeviceToDevice, stream);
  (void)hipMemcpyAsync(q_pos,  iq_pos,  (size_t)BP * 3 * 4, hipMemcpyDeviceToDevice, stream);

  const float scale = 0.17677669529663687f; // 1/sqrt(32)

  for (int l = 0; l < L_; l++) {
    // ----- pe_q MLP -----
    cvt(q_pos, qpos_bf, BP * 3);
    gemm(qpos_bf, 3, 0, bw_qpos1 + (long)l * C_ * 3, 3, 0,
         b_qpos1 + l * C_, nullptr, 0, 0, nullptr, hidq_bf, C_, 0,
         (int)BP, C_, 3, 1, 0, 1);
    gemm(hidq_bf, C_, 0, bw_qpos2 + (long)l * C_ * C_, C_, 0,
         b_qpos2 + l * C_, nullptr, 0, 0, pe_q, nullptr, C_, 0,
         (int)BP, C_, C_, 0, 0, 1);
    // ----- pe_k MLP, kv = keys + pe_k -----
    gemm(cam_bf, 96, 0, bw_kpos1 + (long)l * C_ * 96, 96, 0,
         b_kpos1 + l * C_, nullptr, 0, 0, nullptr, hidk_bf, C_, 0,
         (int)BVN, C_, 96, 1, 0, 1);
    gemm(hidk_bf, C_, 0, bw_kpos2 + (long)l * C_ * C_, C_, 0,
         b_kpos2 + l * C_, keys, C_, 0, nullptr, kv_bf, C_, 0,
         (int)BVN, C_, C_, 0, 0, 1);
    // ----- self-attention -----
    k_broadcast_x<<<ewg(BVP * C_), 256, 0, stream>>>(q_feat, x);
    k_sa_in<<<ewg(BP * C_), 256, 0, stream>>>(q_feat, pe_q, sain_bf);
    gemm(sain_bf, C_, 0, bw_sain + (long)l * 3 * C_ * C_, C_, 0,
         b_sain + l * 3 * C_, nullptr, 0, 0, nullptr, saqkv_bf, 3 * C_, 0,
         (int)BP, 3 * C_, C_, 0, 0, 1);
    for (int b = 0; b < B_; b++) {
      gemm(saqkv_bf + (long)b * P_ * 3 * C_, 3 * C_, 32,
           saqkv_bf + (long)b * P_ * 3 * C_ + C_, 3 * C_, 32,
           nullptr, nullptr, 0, 0,
           lg_sa + (long)b * H_ * P_ * P_, nullptr, P_, (long)P_ * P_,
           P_, P_, 32, 0, 0, H_);
    }
    k_softmax_sa<<<ewg((long)B_ * V_ * H_ * P_ * 32), 256, 0, stream>>>(
        lg_sa, active, pr_sa, scale);
    for (int bv = 0; bv < B_ * V_; bv++) {
      int b = bv / V_;
      gemm(pr_sa + (long)bv * H_ * P_ * P_, P_, (long)P_ * P_,
           saqkv_bf + (long)b * P_ * 3 * C_ + 2 * C_, 3 * C_, 32,
           nullptr, nullptr, 0, 0,
           attnO + (long)bv * H_ * P_ * D_, nullptr, D_, (long)P_ * D_,
           P_, D_, P_, 0, 1, H_);
    }
    k_merge_heads<<<ewg(BVP * C_), 256, 0, stream>>>(attnO, merged);
    gemm(merged, C_, 0, bw_saout + (long)l * C_ * C_, C_, 0,
         b_saout + l * C_, nullptr, 0, 0, projf, nullptr, C_, 0,
         (int)BVP, C_, C_, 0, 0, 1);
    k_ln_res<<<dim3((unsigned)BVP), 256, 0, stream>>>(x, projf, n1g + l * C_, n1b + l * C_, (u16t*)nullptr);
    // ----- cross-attention -----
    k_ca_qin<<<ewg(BVP * C_), 256, 0, stream>>>(x, pe_q, qin_bf);
    gemm(qin_bf, C_, 0, bw_cain + (long)l * 3 * C_ * C_, C_, 0,
         b_cain + l * 3 * C_, nullptr, 0, 0, nullptr, qproj_bf, C_, 0,
         (int)BVP, C_, C_, 0, 0, 1);
    gemm(kv_bf, C_, 0, bw_cain + (long)l * 3 * C_ * C_ + (long)C_ * C_, C_, 0,
         b_cain + l * 3 * C_ + C_, nullptr, 0, 0, nullptr, kproj_bf, C_, 0,
         (int)BVN, C_, C_, 0, 0, 1);
    gemm(kv_bf, C_, 0, bw_cain + (long)l * 3 * C_ * C_ + (long)2 * C_ * C_, C_, 0,
         b_cain + l * 3 * C_ + 2 * C_, nullptr, 0, 0, nullptr, vproj_bf, C_, 0,
         (int)BVN, C_, C_, 0, 0, 1);
    for (int bv = 0; bv < B_ * V_; bv++) {
      gemm(qproj_bf + (long)bv * P_ * C_, C_, 32,
           kproj_bf + (long)bv * N_ * C_, C_, 32,
           nullptr, nullptr, 0, 0,
           lg_ca, nullptr, N_, (long)P_ * N_,
           P_, N_, 32, 0, 0, H_);
      k_softmax_rows<<<ewg((long)H_ * P_ * 32), 256, 0, stream>>>(
          lg_ca, pr_ca, H_ * P_, N_, scale);
      gemm(pr_ca, N_, (long)P_ * N_,
           vproj_bf + (long)bv * N_ * C_, C_, 32,
           nullptr, nullptr, 0, 0,
           attnO + (long)bv * H_ * P_ * D_, nullptr, D_, (long)P_ * D_,
           P_, D_, N_, 0, 1, H_);
    }
    k_merge_heads<<<ewg(BVP * C_), 256, 0, stream>>>(attnO, merged);
    gemm(merged, C_, 0, bw_caout + (long)l * C_ * C_, C_, 0,
         b_caout + l * C_, nullptr, 0, 0, projf, nullptr, C_, 0,
         (int)BVP, C_, C_, 0, 0, 1);
    k_ln_res<<<dim3((unsigned)BVP), 256, 0, stream>>>(x, projf, n2g + l * C_, n2b + l * C_, x_bf);
    // ----- FFN -----
    gemm(x_bf, C_, 0, bw_ffn1 + (long)l * F_ * C_, C_, 0,
         b_ffn1 + l * F_, nullptr, 0, 0, nullptr, hidk_bf, F_, 0,
         (int)BVP, F_, C_, 1, 0, 1);
    gemm(hidk_bf, F_, 0, bw_ffn2 + (long)l * C_ * F_, F_, 0,
         b_ffn2 + l * C_, nullptr, 0, 0, projf, nullptr, C_, 0,
         (int)BVP, C_, F_, 0, 0, 1);
    k_ln_res<<<dim3((unsigned)BVP), 256, 0, stream>>>(x, projf, n3g + l * C_, n3b + l * C_, (u16t*)nullptr);
    // ----- pooling + box head -----
    k_pool<<<ewg(BP * C_), 256, 0, stream>>>(x, active, cnt, q_feat, qT_bf);
    gemm(qT_bf, C_, 0, bw_h1, C_, 0,
         b_h1, nullptr, 0, 0, nullptr, h1_bf, C_, 0,
         (int)BP, C_, C_, 1, 0, 1);
    gemm(h1_bf, C_, 0, bw_h2, C_, 0,
         b_h2, q_pos, 3, 0, center, nullptr, 3, 0,
         (int)BP, 3, C_, 0, 0, 1);
    k_head_finish<<<2, 256, 0, stream>>>(center, q_pos,
                                         out + 104400, out + 106000,
                                         (l == L_ - 1) ? (out + 103600) : nullptr, l);
  }
  (void)hipMemcpyAsync(out, q_feat, (size_t)B_ * C_ * P_ * 4, hipMemcpyDeviceToDevice, stream);
  (void)hipMemcpyAsync(out + 102400, q_pos, (size_t)BP * 3 * 4, hipMemcpyDeviceToDevice, stream);
}
